// DistanceAttentionPerPosition_38293928411601
// MI455X (gfx1250) — matrix operand order repaired and checked by reading, not compiled, not run on hardware
//
#include <hip/hip_runtime.h>
#include <math.h>

typedef __attribute__((ext_vector_type(16))) _Float16 v16h;
typedef __attribute__((ext_vector_type(8)))  float    v8f;

// Problem constants (from reference)
constexpr int N_ATOM = 100;      // padding idx
constexpr int E      = 32;       // EMBED_DIM
constexpr int H      = 256;      // HIDDEN_DIM
constexpr int HA     = 64;       // HIDDEN_DIM/4
constexpr int BB     = 16;
constexpr int WW     = 256;
constexpr int DD     = 64;
constexpr int KIN    = 96;       // 2E+1 = 65 padded up to 3 K-tiles of 32

// LDS strides (in elements) — padded for bank spread, 16B-aligned rows
constexpr int SX = 104;          // X rows (halfs):   104*2 = 208 = 16*13
constexpr int SH = 264;          // hidden rows:      264*2 = 528 = 16*33
constexpr int SA = 66;           // a_gelu rows (f32)

// Workspace layout in halfs: pre-transposed f16 weights, K contiguous
constexpr int WS_WIN   = 0;
constexpr int WS_WA1   = WS_WIN + H * KIN;    // 24576
constexpr int WS_WV1   = WS_WA1 + HA * H;     // 40960
constexpr int WS_WV2   = WS_WV1 + H * H;      // 106496
constexpr int WS_TOTAL = WS_WV2 + H * H;      // 172032 halfs = 344 KB

// ---- branchless transcendentals (single v_exp_f32 / v_rcp_f32 trans ops) ----
#define LOG2E 1.4426950408889634f

__device__ __forceinline__ float fast_exp(float x) {      // e^x, branchless
    return __builtin_amdgcn_exp2f(x * LOG2E);
}
// erf via Abramowitz-Stegun 7.1.26 (|err| < 1.5e-7), fully branchless
__device__ __forceinline__ float fast_erf(float x) {
    float ax = __builtin_fabsf(x);
    float t  = __builtin_amdgcn_rcpf(__builtin_fmaf(0.3275911f, ax, 1.0f));
    float p  = 1.061405429f;
    p = __builtin_fmaf(p, t, -1.453152027f);
    p = __builtin_fmaf(p, t,  1.421413741f);
    p = __builtin_fmaf(p, t, -0.284496736f);
    p = __builtin_fmaf(p, t,  0.254829592f);
    p = p * t;
    float e = __builtin_amdgcn_exp2f(-ax * ax * LOG2E);
    float r = __builtin_fmaf(-p, e, 1.0f);
    return __builtin_copysignf(r, x);
}
__device__ __forceinline__ float gelu_fast(float x) {
    return 0.5f * x * (1.0f + fast_erf(x * 0.70710678118654752f));
}

union FragU { v16h v; float4 f[2]; };
union Half8 { float4 f; _Float16 h[8]; };

// A fragment (16x32 f16): lane = 16*h + m(0..15); element i<8 -> k=kb+8h+i,
// element i>=8 -> k=kb+8h+16+(i-8). Row pointer is the M row base.
__device__ __forceinline__ v16h load_a_frag(const _Float16* row, int kb, int h) {
    FragU a;
    a.f[0] = *(const float4*)(row + kb + 8 * h);
    a.f[1] = *(const float4*)(row + kb + 8 * h + 16);
    return a.v;
}

// B fragment (32x16 f16): lane = 16*h + n(0..15); element i<8 -> k=kb+8h+i,
// element i>=8 -> k=kb+8h+16+(i-8).
// colrow points at the K-contiguous transposed weight row for column n.
__device__ __forceinline__ v16h load_b_frag(const _Float16* colrow, int kb, int h) {
    FragU b;
    const _Float16* p = colrow + kb + 8 * h;
    b.f[0] = *(const float4*)(p);
    b.f[1] = *(const float4*)(p + 16);
    return b.v;
}

// One-shot weight preprocessing: f32 [K][N] -> f16 [N][K] (K contiguous, padded)
__global__ void prep_weights_kernel(const float* __restrict__ w_in,
                                    const float* __restrict__ w_a1,
                                    const float* __restrict__ w_v1,
                                    const float* __restrict__ w_v2,
                                    _Float16* __restrict__ ws) {
    int t = blockIdx.x * blockDim.x + threadIdx.x;
    if (t >= WS_TOTAL) return;
    float v;
    if (t < WS_WA1) {                       // w_inT [256][96], K=65 real
        int r = t - WS_WIN; int n = r / KIN, k = r - n * KIN;
        v = (k < 2 * E + 1) ? w_in[k * H + n] : 0.0f;
    } else if (t < WS_WV1) {                // w_a1T [64][256]
        int r = t - WS_WA1; int n = r / H, k = r - n * H;
        v = w_a1[k * HA + n];
    } else if (t < WS_WV2) {                // w_v1T [256][256]
        int r = t - WS_WV1; int n = r / H, k = r - n * H;
        v = w_v1[k * H + n];
    } else {                                // w_v2T [256][256]
        int r = t - WS_WV2; int n = r / H, k = r - n * H;
        v = w_v2[k * H + n];
    }
    ws[t] = (_Float16)v;
}

__global__ __launch_bounds__(256)
void fused_distance_attention(const int* __restrict__ a1i,
                              const int* __restrict__ a2i,
                              const float* __restrict__ dist,
                              const unsigned char* __restrict__ mask,
                              const float* __restrict__ embed,
                              const float* __restrict__ b_in,
                              const float* __restrict__ b_a1,
                              const float* __restrict__ w_a2,
                              const float* __restrict__ b_a2,
                              const float* __restrict__ b_v1,
                              const float* __restrict__ b_v2,
                              const float* __restrict__ fallback,
                              const _Float16* __restrict__ ws,
                              float* __restrict__ out) {
    __shared__ _Float16 hs[DD * SH];          // hidden, f16   (33792 B)
    __shared__ float    uni[DD * SA];         // X overlay then a_gelu (16896 B)
    __shared__ float    scores_s[DD];
    __shared__ float    attn_s[DD];
    __shared__ float    mask_s[DD];
    __shared__ float    z_s[H];
    __shared__ int      anyv_s;

    _Float16* Xs  = (_Float16*)uni;           // [DD][SX] halfs (13312 B, fits)
    float*    as_ = uni;                      // [DD][SA] floats

    const int tid  = threadIdx.x;
    const int lane = tid & 31;
    const int wv   = tid >> 5;                // wave id 0..7
    const int hh   = lane >> 4;               // lane half
    const int ln   = lane & 15;
    const int bw   = blockIdx.x;              // 0 .. B*W-1
    const long base = (long)bw * DD;

    // ---------------- Phase 0: build X = [e1 | e2 | d*mask | 0pad] in f16 LDS
    for (int e = tid; e < DD * KIN; e += 256) {
        int m = e / KIN, k = e - m * KIN;
        float v = 0.0f;
        if (k < E) {
            int idx = a1i[base + m];
            v = embed[idx * E + k] * (idx != N_ATOM ? 1.0f : 0.0f);
        } else if (k < 2 * E) {
            int idx = a2i[base + m];
            v = embed[idx * E + (k - E)] * (idx != N_ATOM ? 1.0f : 0.0f);
        } else if (k == 2 * E) {
            float mk = mask[base + m] ? 1.0f : 0.0f;
            v = dist[base + m] * mk;
        }
        Xs[m * SX + k] = (_Float16)v;
    }
    if (tid < DD) mask_s[tid] = mask[base + tid] ? 1.0f : 0.0f;
    if (tid == 0) anyv_s = 0;
    __syncthreads();

    // ---------------- Phase 1: hidden = gelu(X @ w_in + b_in), WMMA, K=96
    // K outermost: A-fragments loaded once per K-step, shared by both N-tiles.
    {
        const _Float16* wInT = ws + WS_WIN;
        const int n0 = wv * 32;                        // 32 columns per wave
        v8f acc[2][4] = {};
#pragma unroll
        for (int kt = 0; kt < 3; ++kt) {
            v16h afr[4];
#pragma unroll
            for (int mt = 0; mt < 4; ++mt)
                afr[mt] = load_a_frag(Xs + (mt * 16 + ln) * SX, kt * 32, hh);
#pragma unroll
            for (int nt = 0; nt < 2; ++nt) {
                v16h bfr = load_b_frag(wInT + (n0 + nt * 16 + ln) * KIN, kt * 32, hh);
#pragma unroll
                for (int mt = 0; mt < 4; ++mt)
                    acc[nt][mt] = __builtin_amdgcn_wmma_f32_16x16x32_f16(
                        false, afr[mt], false, bfr, (short)0, acc[nt][mt], false, false);
            }
        }
#pragma unroll
        for (int nt = 0; nt < 2; ++nt) {
            const float bn = b_in[n0 + nt * 16 + ln];
#pragma unroll
            for (int mt = 0; mt < 4; ++mt)
#pragma unroll
                for (int j = 0; j < 8; ++j) {
                    int m = mt * 16 + j + 8 * hh;       // D layout: VGPR j, half hh
                    hs[m * SH + n0 + nt * 16 + ln] =
                        (_Float16)gelu_fast(acc[nt][mt][j] + bn);
                }
        }
    }
    __syncthreads();

    // ---------------- Phase 2: a = gelu(hidden @ w_a1 + b_a1) -> [64][64] f32
    {
        const _Float16* wA1T = ws + WS_WA1;
#pragma unroll
        for (int i = 0; i < 2; ++i) {                  // 2 tiles per wave
            const int t  = wv * 2 + i;
            const int mt = t >> 2, nt2 = t & 3;
            v8f acc = {};
#pragma unroll
            for (int kt = 0; kt < 8; ++kt) {
                v16h bfr = load_b_frag(wA1T + (nt2 * 16 + ln) * H, kt * 32, hh);
                v16h afr = load_a_frag(hs + (mt * 16 + ln) * SH, kt * 32, hh);
                acc = __builtin_amdgcn_wmma_f32_16x16x32_f16(
                    false, afr, false, bfr, (short)0, acc, false, false);
            }
            const float bn = b_a1[nt2 * 16 + ln];
#pragma unroll
            for (int j = 0; j < 8; ++j) {
                int m = mt * 16 + j + 8 * hh;
                as_[m * SA + nt2 * 16 + ln] = gelu_fast(acc[j] + bn);
            }
        }
    }
    __syncthreads();

    // ---------------- Phase 2b: scores[m] = a[m]·w_a2 + b_a2, mask -> -1e4
    if (tid < DD) {
        float s = 0.0f;
#pragma unroll 8
        for (int j = 0; j < HA; ++j) s += as_[tid * SA + j] * w_a2[j];
        s += b_a2[0];
        scores_s[tid] = (mask_s[tid] != 0.0f) ? s : -10000.0f;
    }
    if (tid == 0) {
        int anyv = 0;
        for (int m = 0; m < DD; ++m) anyv |= (mask_s[m] != 0.0f);
        anyv_s = anyv;
    }
    __syncthreads();

    // ---------------- Phase 3: softmax over D (redundant read, tid<64 writes)
    {
        float mx = -1e30f;
        for (int m = 0; m < DD; ++m) mx = fmaxf(mx, scores_s[m]);
        float den = 0.0f;
        for (int m = 0; m < DD; ++m) den += fast_exp(scores_s[m] - mx);
        float rden = __builtin_amdgcn_rcpf(den);
        if (tid < DD) attn_s[tid] = fast_exp(scores_s[tid] - mx) * rden;
    }
    __syncthreads();

    // ---------------- Phase 4: z = attn^T @ gelu(hidden @ w_v1 + b_v1)
    // (fused: weight-and-reduce in the WMMA epilogue; no v1 staging needed
    //  since out = (attn^T @ v1) @ w_v2 + b_v2, softmax sums to 1)
    // K outermost: A-fragments loaded once per K-step, shared by both N-tiles.
    {
        const _Float16* wV1T = ws + WS_WV1;
        const int n0 = wv * 32;
        v8f acc[2][4] = {};
#pragma unroll
        for (int kt = 0; kt < 8; ++kt) {
            v16h afr[4];
#pragma unroll
            for (int mt = 0; mt < 4; ++mt)
                afr[mt] = load_a_frag(hs + (mt * 16 + ln) * SH, kt * 32, hh);
#pragma unroll
            for (int nt = 0; nt < 2; ++nt) {
                v16h bfr = load_b_frag(wV1T + (n0 + nt * 16 + ln) * H, kt * 32, hh);
#pragma unroll
                for (int mt = 0; mt < 4; ++mt)
                    acc[nt][mt] = __builtin_amdgcn_wmma_f32_16x16x32_f16(
                        false, afr[mt], false, bfr, (short)0, acc[nt][mt], false, false);
            }
        }
#pragma unroll
        for (int nt = 0; nt < 2; ++nt) {
            const float bn = b_v1[n0 + nt * 16 + ln];
            float zp = 0.0f;
#pragma unroll
            for (int mt = 0; mt < 4; ++mt)
#pragma unroll
                for (int j = 0; j < 8; ++j) {
                    int m = mt * 16 + j + 8 * hh;
                    zp = __builtin_fmaf(attn_s[m], gelu_fast(acc[nt][mt][j] + bn), zp);
                }
            // combine the two lane-halves (each covers half the M rows)
            float tot = zp + __shfl_xor(zp, 16, 32);
            if (hh == 0) z_s[n0 + nt * 16 + ln] = tot;
        }
    }
    __syncthreads();

    // ---------------- Phase 5: out = z @ w_v2 + b_v2 ; fallback if no valid
    {
        const _Float16* row = ws + WS_WV2 + tid * H;   // w_v2T row for column tid
        float accv = b_v2[tid];
        for (int k8 = 0; k8 < H; k8 += 8) {
            Half8 u;
            u.f = *(const float4*)(row + k8);          // 8 halfs per b128 load
#pragma unroll
            for (int j = 0; j < 8; ++j)
                accv = __builtin_fmaf(z_s[k8 + j], (float)u.h[j], accv);
        }
        out[(long)bw * H + tid] = anyv_s ? accv : fallback[tid];
    }
}

extern "C" void kernel_launch(void* const* d_in, const int* in_sizes, int n_in,
                              void* d_out, int out_size, void* d_ws, size_t ws_size,
                              hipStream_t stream) {
    const int*           a1       = (const int*)d_in[0];
    const int*           a2       = (const int*)d_in[1];
    const float*         dist     = (const float*)d_in[2];
    const unsigned char* mask     = (const unsigned char*)d_in[3];  // jax bool = 1B
    const float*         embed    = (const float*)d_in[4];
    const float*         w_in     = (const float*)d_in[5];
    const float*         b_in     = (const float*)d_in[6];
    const float*         w_a1     = (const float*)d_in[7];
    const float*         b_a1     = (const float*)d_in[8];
    const float*         w_a2     = (const float*)d_in[9];
    const float*         b_a2     = (const float*)d_in[10];
    const float*         w_v1     = (const float*)d_in[11];
    const float*         b_v1     = (const float*)d_in[12];
    const float*         w_v2     = (const float*)d_in[13];
    const float*         b_v2     = (const float*)d_in[14];
    const float*         fallback = (const float*)d_in[15];
    _Float16*            ws       = (_Float16*)d_ws;
    float*               out      = (float*)d_out;

    (void)in_sizes; (void)n_in; (void)out_size; (void)ws_size;

    prep_weights_kernel<<<(WS_TOTAL + 255) / 256, 256, 0, stream>>>(
        w_in, w_a1, w_v1, w_v2, ws);

    fused_distance_attention<<<BB * WW, 256, 0, stream>>>(
        a1, a2, dist, mask, embed, b_in, b_a1, w_a2, b_a2,
        b_v1, b_v2, fallback, ws, out);
}
